// AttentionHead_31774168056253
// MI455X (gfx1250) — matrix operand order repaired and checked by reading, not compiled, not run on hardware
//
#include <hip/hip_runtime.h>
#include <hip/hip_bf16.h>

// AttentionHead: out = softmax(mask((X@Q)(X@K)^T / sqrt(D))) @ (X@V)
// B=8, S=2048, E=768, D=64.
// bf16 WMMA (v_wmma_f32_16x16x32_bf16) for all GEMMs; flash attention
// (never materialize SxS); X tile staged in LDS via async-to-LDS (one load
// per block instead of 4x redundant per-wave loads); weights pre-swizzled
// into B-fragment dword layout; V emitted row-pair-packed for coalesced
// B-fragment loads in the attention kernel.

static constexpr int E  = 768;
static constexpr int D  = 64;
static constexpr int Bb = 8;
static constexpr int S  = 2048;

typedef __attribute__((ext_vector_type(16))) __bf16 v16bf;
typedef __attribute__((ext_vector_type(2)))  __bf16 v2bf;
typedef __attribute__((ext_vector_type(8)))  float  v8f;
typedef __attribute__((ext_vector_type(4)))  int    v4i;

union BF16x16 {
    v16bf v;
    unsigned short s[16];
    unsigned int   u[8];
};

// native f32->bf16 (hardware RNE cvt on gfx1250)
__device__ __forceinline__ unsigned short f2bf(float f) {
    union { __bf16 b; unsigned short s; } c;
    c.b = (__bf16)f;
    return c.s;
}
__device__ __forceinline__ unsigned int pack2bf(float lo, float hi) {
    union { v2bf v; unsigned int u; } c;
    c.v.x = (__bf16)lo;
    c.v.y = (__bf16)hi;
    return c.u;
}
// 16-bit A-matrix (16x32) K index for VGPR v, lane-half h (ISA 7.12.2)
__device__ __forceinline__ int aK(int v, int h) {
    return (v < 4) ? (2 * v + 8 * h) : (16 + 2 * (v - 4) + 8 * h);
}

#if defined(__HIP_DEVICE_COMPILE__) && __has_builtin(__builtin_amdgcn_global_load_async_to_lds_b128)
#define ASYNC_LDS 1
#else
#define ASYNC_LDS 0
#endif

// ---------------------------------------------------------------------------
// Kernel 0: pre-swizzle weights fp32[E][D] -> bf16 row-pair-packed dwords:
// Wp[p*D + n] = {W[2p][n], W[2p+1][n]}  (exactly one dword per B-fragment VGPR)
// ---------------------------------------------------------------------------
__global__ __launch_bounds__(256) void swizzle_w(
        const float* __restrict__ Wq, const float* __restrict__ Wk,
        const float* __restrict__ Wv,
        unsigned int* __restrict__ Wqp, unsigned int* __restrict__ Wkp,
        unsigned int* __restrict__ Wvp) {
    int tid = blockIdx.x * 256 + threadIdx.x;        // over (E/2)*D
    if (tid >= (E / 2) * D) return;
    int p = tid / D, n = tid % D;
    size_t a0 = (size_t)(2 * p) * D + n;
    Wqp[tid] = pack2bf(Wq[a0], Wq[a0 + D]);
    Wkp[tid] = pack2bf(Wk[a0], Wk[a0 + D]);
    Wvp[tid] = pack2bf(Wv[a0], Wv[a0 + D]);
}

// ---------------------------------------------------------------------------
// Kernel 1: q/k/v = X @ {Q,K,V}.  Grid: Bb*(S/16) blocks of 128 threads.
// X tile (16x32 f32, 2 KB) staged in LDS once per block (async-to-LDS when
// available).  Wave w owns output N-tile w.  q,k stored plain bf16; v stored
// row-pair-packed along S for the attention kernel's B fragments.
// ---------------------------------------------------------------------------
__global__ __launch_bounds__(128) void proj_qkv(
        const float* __restrict__ in,
        const unsigned int* __restrict__ Wqp, const unsigned int* __restrict__ Wkp,
        const unsigned int* __restrict__ Wvp,
        unsigned short* __restrict__ qb, unsigned short* __restrict__ kb,
        unsigned int* __restrict__ vp) {
    __shared__ float Xt[16 * 32];

    const int wave = threadIdx.x >> 5;
    const int lane = threadIdx.x & 31;
    const int half = lane >> 4;
    const int lm   = lane & 15;
    const int bI   = blockIdx.x / (S / 16);
    const int q0   = (blockIdx.x % (S / 16)) * 16;
    const int n    = wave * 16 + lm;                 // output column (head dim)

    // cooperative stager: thread t covers row t/8, 4 cols at (t%8)*4
    const int trow = threadIdx.x >> 3;
    const int tcol = (threadIdx.x & 7) * 4;
    const float* srcrow = in + ((size_t)bI * S + q0 + trow) * E + tcol;

    v8f aq = {}, ak = {}, av = {};
#pragma unroll 1
    for (int ks = 0; ks < E / 32; ++ks) {
        // ---- stage X tile into LDS (one 16B transfer per thread)
#if ASYNC_LDS
        __builtin_amdgcn_global_load_async_to_lds_b128(
            (v4i*)(srcrow + ks * 32), (v4i*)&Xt[trow * 32 + tcol], 0, 0);
#if __has_builtin(__builtin_amdgcn_s_wait_asynccnt)
        __builtin_amdgcn_s_wait_asynccnt(0);
#else
        asm volatile("s_wait_asynccnt 0" ::: "memory");
#endif
#else
        *(float4*)&Xt[trow * 32 + tcol] = *(const float4*)(srcrow + ks * 32);
#endif
        __syncthreads();

        // ---- A fragment (16x32, row = lm) from LDS, cvt f32->bf16 pairs
        BF16x16 A;
#pragma unroll
        for (int v = 0; v < 8; ++v) {
            const float* p = &Xt[lm * 32 + 16 * half + 2 * v];
            A.u[v] = pack2bf(p[0], p[1]);
        }
        // ---- B fragments: one coalesced dword per VGPR from swizzled weights
        size_t wb = (size_t)(ks * 16 + 8 * half) * D + n;
        BF16x16 Bq, Bk, Bv;
#pragma unroll
        for (int v = 0; v < 8; ++v) {
            Bq.u[v] = Wqp[wb + (size_t)v * D];
            Bk.u[v] = Wkp[wb + (size_t)v * D];
            Bv.u[v] = Wvp[wb + (size_t)v * D];
        }
        aq = __builtin_amdgcn_wmma_f32_16x16x32_bf16(false, A.v, false, Bq.v, (short)0, aq, false, false);
        ak = __builtin_amdgcn_wmma_f32_16x16x32_bf16(false, A.v, false, Bk.v, (short)0, ak, false, false);
        av = __builtin_amdgcn_wmma_f32_16x16x32_bf16(false, A.v, false, Bv.v, (short)0, av, false, false);
        __syncthreads();
    }

    // ---- store q,k plain (C-layout: VGPR r -> rows r / r+8, col n)
#pragma unroll
    for (int r = 0; r < 8; ++r) {
        size_t o = ((size_t)bI * S + q0 + r + 8 * half) * D + n;
        qb[o] = f2bf(aq[r]);
        kb[o] = f2bf(ak[r]);
    }
    // ---- store v row-pair-packed along S: pairs (2i,2i+1) live in-lane
#pragma unroll
    for (int i = 0; i < 4; ++i) {
        size_t o = (((size_t)bI * S + q0) / 2 + 4 * half + i) * D + n;
        vp[o] = pack2bf(av[2 * i], av[2 * i + 1]);
    }
}

// ---------------------------------------------------------------------------
// Kernel 2: causal flash attention.  One wave per 16-query tile; key blocks
// of 32 (two 16x16 S tiles) so P@V uses full K=32 WMMAs.
// Grid: Bb*(S/64) blocks of 128 threads (4 independent waves).
// ---------------------------------------------------------------------------
__global__ __launch_bounds__(128) void flash_attn(
        const unsigned short* __restrict__ qb,
        const unsigned short* __restrict__ kb,
        const unsigned int*  __restrict__ vp,
        float* __restrict__ out) {
    __shared__ unsigned short Plds[4][16 * 32];      // per-wave P scratch (bf16)

    const int wave = threadIdx.x >> 5;
    const int lane = threadIdx.x & 31;
    const int half = lane >> 4;
    const int lm   = lane & 15;
    const int bI   = blockIdx.x / (S / 64);
    const int qt   = (blockIdx.x % (S / 64)) * 4 + wave;
    const int q0   = qt * 16;

    const unsigned int* qb32 = (const unsigned int*)qb;
    const unsigned int* kb32 = (const unsigned int*)kb;

    // Q A-fragments for head-dim K-steps 0..31 and 32..63 (held all loop)
    BF16x16 Af[2];
    {
        size_t base = ((size_t)bI * S + q0 + lm) * (D / 2);
#pragma unroll
        for (int s = 0; s < 2; ++s)
#pragma unroll
            for (int v = 0; v < 8; ++v)
                Af[s].u[v] = qb32[base + (s * 32 + 16 * half + 2 * v) / 2];
    }

    float mrow[8], lrow[8], alpha[8];
    v8f o0 = {}, o1 = {}, o2 = {}, o3 = {};
#pragma unroll
    for (int r = 0; r < 8; ++r) { mrow[r] = -__builtin_inff(); lrow[r] = 0.f; }

    const float scale = 0.125f;                      // 1/sqrt(64)
    const int jmax = (q0 + 15) / 32;
    for (int j = 0; j <= jmax; ++j) {
        if (j < jmax) {                              // prefetch next key block
            __builtin_prefetch(&kb[((size_t)bI * S + 32 * (j + 1) + lm) * D], 0, 1);
            __builtin_prefetch(&vp[((size_t)bI * S / 2 + 16 * (j + 1)) * D + lm], 0, 1);
        }

        // ---- S = q @ k^T : two 16x16 tiles (keys 32j..+15 and +16..+31)
        v8f c0 = {}, c1 = {};
#pragma unroll
        for (int t = 0; t < 2; ++t) {
            int key = 32 * j + 16 * t + lm;          // B column = key position
            size_t kbase = ((size_t)bI * S + key) * (D / 2);
            v8f c = {};
#pragma unroll
            for (int s = 0; s < 2; ++s) {
                BF16x16 Bf;                          // B 32x16: rows = head dims
#pragma unroll
                for (int v = 0; v < 8; ++v)
                    Bf.u[v] = kb32[kbase + (s * 32 + 2 * v + 16 * half) / 2];
                c = __builtin_amdgcn_wmma_f32_16x16x32_bf16(false, Af[s].v, false, Bf.v, (short)0, c, false, false);
            }
            if (t == 0) c0 = c; else c1 = c;
        }

        // ---- scale + causal mask + online softmax (rows in 16-lane halves)
#pragma unroll
        for (int r = 0; r < 8; ++r) {
            int M  = q0 + r + 8 * half;
            int k0 = 32 * j + lm;
            float s0 = c0[r] * scale; if (k0 > M)      s0 = -1e30f;
            float s1 = c1[r] * scale; if (k0 + 16 > M) s1 = -1e30f;
            float mx = fmaxf(s0, s1);
#pragma unroll
            for (int d = 1; d < 16; d <<= 1) mx = fmaxf(mx, __shfl_xor(mx, d, 16));
            float mn = fmaxf(mrow[r], mx);
            alpha[r] = __expf(mrow[r] - mn);
            float p0 = __expf(s0 - mn);
            float p1 = __expf(s1 - mn);
            float rs = p0 + p1;
#pragma unroll
            for (int d = 1; d < 16; d <<= 1) rs += __shfl_xor(rs, d, 16);
            lrow[r] = lrow[r] * alpha[r] + rs;
            mrow[r] = mn;
            int Ml = r + 8 * half;                   // write P (C-layout -> LDS)
            Plds[wave][Ml * 32 + lm]      = f2bf(p0);
            Plds[wave][Ml * 32 + 16 + lm] = f2bf(p1);
        }
#pragma unroll
        for (int r = 0; r < 8; ++r) {                // rescale O accumulators
            o0[r] *= alpha[r]; o1[r] *= alpha[r];
            o2[r] *= alpha[r]; o3[r] *= alpha[r];
        }

        // ---- reload P as A-fragment (16x32, row = lm) from LDS
        BF16x16 Pf;
        {
            const unsigned int* pl = (const unsigned int*)&Plds[wave][0];
#pragma unroll
            for (int v = 0; v < 8; ++v)
                Pf.u[v] = pl[lm * 16 + 8 * half + v];
        }

        // ---- O += P @ V : four 16x16 output tiles over D=64
        size_t vbase = ((size_t)bI * S / 2 + 16 * j) * D;
#pragma unroll
        for (int t = 0; t < 4; ++t) {
            BF16x16 Vf;                              // one coalesced dword per VGPR
#pragma unroll
            for (int v = 0; v < 8; ++v)
                Vf.u[v] = vp[vbase + (size_t)(v + 8 * half) * D + t * 16 + lm];
            v8f& oacc = (t == 0 ? o0 : t == 1 ? o1 : t == 2 ? o2 : o3);
            oacc = __builtin_amdgcn_wmma_f32_16x16x32_bf16(false, Pf.v, false, Vf.v, (short)0, oacc, false, false);
        }
    }

    // ---- normalize and store fp32 output [B,S,D]
#pragma unroll
    for (int r = 0; r < 8; ++r) {
        float inv = 1.0f / lrow[r];
        size_t ob = ((size_t)bI * S + q0 + r + 8 * half) * D + lm;
        out[ob +  0] = o0[r] * inv;
        out[ob + 16] = o1[r] * inv;
        out[ob + 32] = o2[r] * inv;
        out[ob + 48] = o3[r] * inv;
    }
}

// ---------------------------------------------------------------------------
extern "C" void kernel_launch(void* const* d_in, const int* in_sizes, int n_in,
                              void* d_out, int out_size, void* d_ws, size_t ws_size,
                              hipStream_t stream) {
    const float* in = (const float*)d_in[0];
    // d_in[1] = attention_mask (causal tril) -- implemented implicitly
    const float* Wq = (const float*)d_in[2];
    const float* Wk = (const float*)d_in[3];
    const float* Wv = (const float*)d_in[4];

    const size_t N = (size_t)Bb * S * D;             // per-tensor elements
    unsigned short* qb  = (unsigned short*)d_ws;     // 2 MB bf16
    unsigned short* kb  = qb + N;                    // 2 MB bf16
    unsigned int*   vp  = (unsigned int*)(kb + N);   // 2 MB bf16 (pair-packed)
    unsigned int*   Wqp = vp + N / 2;                // 3 x 96 KB swizzled weights
    unsigned int*   Wkp = Wqp + (E / 2) * D;
    unsigned int*   Wvp = Wkp + (E / 2) * D;

    swizzle_w<<<((E / 2) * D + 255) / 256, 256, 0, stream>>>(Wq, Wk, Wv, Wqp, Wkp, Wvp);
    proj_qkv<<<Bb * (S / 16), 128, 0, stream>>>(in, Wqp, Wkp, Wvp, qb, kb, vp);
    flash_attn<<<Bb * (S / 64), 128, 0, stream>>>(qb, kb, vp, (float*)d_out);
}
